// BLTModel_4320737100317
// MI455X (gfx1250) — compile-verified, matrix operand order checked by reading, not run on hardware
//
#include <hip/hip_runtime.h>
#include <hip/hip_bf16.h>

// ---------------------------------------------------------------------------
// BLT forward on gfx1250 (CDNA5). All matrix math via v_wmma_f32_16x16x32_f16
// (fp32 params converted to f16 at staging, fp32 accumulate).
// Dims: B=4 S=1024 PATCH=8 P=128 DE=768 HE=12 DG=1024 HG=16 DD=768 HD=12
// All GEMM shapes are multiples of 64(M)/128(N)/32(K) -> no bounds guards.
// ---------------------------------------------------------------------------

typedef __attribute__((ext_vector_type(16))) _Float16 v16h;
typedef __attribute__((ext_vector_type(8)))  _Float16 v8h;
typedef __attribute__((ext_vector_type(2)))  _Float16 v2h;
typedef __attribute__((ext_vector_type(8)))  float    v8f;

#define WMMA_F32_F16(a, b, c) \
  __builtin_amdgcn_wmma_f32_16x16x32_f16(false, (a), false, (b), (short)0, (c), false, false)

// k index for element-pair j (0..7) of a 16-bit A/B fragment, per CDNA5 ISA
// 16-bit A-matrix 16x32 layout: lanes 0-15 kbase=0, lanes 16-31 kbase=8;
// VGPR j<4 -> k=2j+kbase, j>=4 -> k=16+2(j-4)+kbase.
__device__ __forceinline__ int frag_k(int j, int hlf) {
  return ((j < 4) ? (2 * j) : (16 + 2 * (j - 4))) + 8 * hlf;
}

// Load a 32-wide f32 row chunk into an f16 A/B-style fragment for this lane.
// Needed k-values are the two contiguous 8-float runs [8*hlf, 8*hlf+7] and
// [16+8*hlf, 16+8*hlf+7] -> four float4 (global_load_b128) loads.
__device__ __forceinline__ v16h load_row_frag(const float* __restrict__ row, int hlf) {
  const float* p = row + 8 * hlf;
  float4 x0 = *(const float4*)(p);
  float4 x1 = *(const float4*)(p + 4);
  float4 x2 = *(const float4*)(p + 16);
  float4 x3 = *(const float4*)(p + 20);
  v16h f;
  f[0]  = (_Float16)x0.x; f[1]  = (_Float16)x0.y;
  f[2]  = (_Float16)x0.z; f[3]  = (_Float16)x0.w;
  f[4]  = (_Float16)x1.x; f[5]  = (_Float16)x1.y;
  f[6]  = (_Float16)x1.z; f[7]  = (_Float16)x1.w;
  f[8]  = (_Float16)x2.x; f[9]  = (_Float16)x2.y;
  f[10] = (_Float16)x2.z; f[11] = (_Float16)x2.w;
  f[12] = (_Float16)x3.x; f[13] = (_Float16)x3.y;
  f[14] = (_Float16)x3.z; f[15] = (_Float16)x3.w;
  return f;
}

// ---------------------------------------------------------------------------
// Tiled WMMA GEMM: C[M,N] = A[M,K] @ W[K,N] (+bias) (+C) (relu)
// A, W, C row-major fp32. flags: bit0 = relu, bit1 = accumulate into C.
// Block: 256 threads (8 waves), 64(M) x 128(N) tile, K step 32.
// Each wave: 32m x 32n -> 4 WMMAs per k-step.
// Requires: M % 64 == 0, N % 128 == 0, K % 32 == 0 (true at all call sites).
// ---------------------------------------------------------------------------
__global__ __launch_bounds__(256)
void gemm_wmma_kernel(const float* __restrict__ A, const float* __restrict__ W,
                      const float* __restrict__ bias, float* __restrict__ C,
                      int M, int N, int K, int flags) {
  __shared__ __align__(16) _Float16 As[64][40];    // [m][k]
  __shared__ __align__(16) _Float16 Bs[128][40];   // [n][k]  (W transposed)
  const int tid  = threadIdx.x;
  const int lane = tid & 31;
  const int wave = tid >> 5;        // 0..7
  const int wr   = wave >> 2;       // m 32-tile 0..1
  const int wc   = wave & 3;        // n 32-tile 0..3
  const int hlf  = lane >> 4;
  const int lm   = lane & 15;
  const int mbase = blockIdx.y * 64;
  const int nbase = blockIdx.x * 128;

  // staging maps (exact, no guards)
  const int arow = tid >> 2;            // 0..63  (m)
  const int aseg = (tid & 3) * 8;       // 0,8,16,24 (k)
  const int bk   = (tid >> 4) * 2;      // 0,2,..,30 (k pair)
  const int bng  = (tid & 15) * 8;      // 0..120 (n)

  const v8f zero8 = {0.f, 0.f, 0.f, 0.f, 0.f, 0.f, 0.f, 0.f};
  v8f acc00 = zero8, acc01 = zero8, acc10 = zero8, acc11 = zero8;

  for (int k0 = 0; k0 < K; k0 += 32) {
    // ---- stage A: 64 x 32, one b128 DS store per thread ----
    {
      const float* ap = A + (size_t)(mbase + arow) * K + k0 + aseg;
      __builtin_prefetch(ap + 32, 0, 0);
      float4 a01 = *(const float4*)ap;
      float4 a23 = *(const float4*)(ap + 4);
      v8h h;
      h[0] = (_Float16)a01.x; h[1] = (_Float16)a01.y;
      h[2] = (_Float16)a01.z; h[3] = (_Float16)a01.w;
      h[4] = (_Float16)a23.x; h[5] = (_Float16)a23.y;
      h[6] = (_Float16)a23.z; h[7] = (_Float16)a23.w;
      *(v8h*)(&As[arow][aseg]) = h;
    }
    // ---- stage B (transposed): thread owns k-pair x 8 n -> 8 packed b32 ----
    {
      const float* wp0 = W + (size_t)(k0 + bk) * N + nbase + bng;
      const float* wp1 = wp0 + N;
      __builtin_prefetch(wp0 + (size_t)32 * N, 0, 0);
      float4 r0a = *(const float4*)wp0;
      float4 r0b = *(const float4*)(wp0 + 4);
      float4 r1a = *(const float4*)wp1;
      float4 r1b = *(const float4*)(wp1 + 4);
      float r0[8] = {r0a.x, r0a.y, r0a.z, r0a.w, r0b.x, r0b.y, r0b.z, r0b.w};
      float r1[8] = {r1a.x, r1a.y, r1a.z, r1a.w, r1b.x, r1b.y, r1b.z, r1b.w};
#pragma unroll
      for (int e = 0; e < 8; ++e) {
        v2h p;
        p[0] = (_Float16)r0[e];
        p[1] = (_Float16)r1[e];
        *(v2h*)(&Bs[bng + e][bk]) = p;
      }
    }
    __syncthreads();
    v16h af0, af1, bf0, bf1;
#pragma unroll
    for (int j = 0; j < 8; ++j) {
      int kk = frag_k(j, hlf);
      af0[2 * j]     = As[wr * 32 + lm][kk];
      af0[2 * j + 1] = As[wr * 32 + lm][kk + 1];
      af1[2 * j]     = As[wr * 32 + 16 + lm][kk];
      af1[2 * j + 1] = As[wr * 32 + 16 + lm][kk + 1];
      bf0[2 * j]     = Bs[wc * 32 + lm][kk];
      bf0[2 * j + 1] = Bs[wc * 32 + lm][kk + 1];
      bf1[2 * j]     = Bs[wc * 32 + 16 + lm][kk];
      bf1[2 * j + 1] = Bs[wc * 32 + 16 + lm][kk + 1];
    }
    acc00 = WMMA_F32_F16(af0, bf0, acc00);
    acc01 = WMMA_F32_F16(af0, bf1, acc01);
    acc10 = WMMA_F32_F16(af1, bf0, acc10);
    acc11 = WMMA_F32_F16(af1, bf1, acc11);
    __syncthreads();
  }

#pragma unroll
  for (int i = 0; i < 8; ++i) {
    int m0 = mbase + wr * 32 + i + 8 * hlf;
    int m1 = m0 + 16;
    int n0 = nbase + wc * 32 + lm;
    int n1 = n0 + 16;
    float v00 = acc00[i], v01 = acc01[i], v10 = acc10[i], v11 = acc11[i];
    if (bias) {
      float b0 = bias[n0], b1 = bias[n1];
      v00 += b0; v01 += b1; v10 += b0; v11 += b1;
    }
    if (flags & 2) {
      v00 += C[(size_t)m0 * N + n0]; v01 += C[(size_t)m0 * N + n1];
      v10 += C[(size_t)m1 * N + n0]; v11 += C[(size_t)m1 * N + n1];
    }
    if (flags & 1) {
      v00 = fmaxf(v00, 0.f); v01 = fmaxf(v01, 0.f);
      v10 = fmaxf(v10, 0.f); v11 = fmaxf(v11, 0.f);
    }
    C[(size_t)m0 * N + n0] = v00;
    C[(size_t)m0 * N + n1] = v01;
    C[(size_t)m1 * N + n0] = v10;
    C[(size_t)m1 * N + n1] = v11;
  }
}

// ---------------------------------------------------------------------------
// Flash attention, dh = 64. One wave per (b, head, 16-query tile).
// Q,K,V,O: [B, T, D] fp32, head h occupies columns h*64..h*64+63.
// Tk is a multiple of 32 at every call site -> no tail guards.
// ---------------------------------------------------------------------------
__global__ __launch_bounds__(32)
void flash_attn_kernel(const float* __restrict__ Q, const float* __restrict__ Kt,
                       const float* __restrict__ Vt, float* __restrict__ O,
                       int Tq, int Tk, int D, int causal) {
  const int DH   = 64;
  const int lane = threadIdx.x;
  const int hlf  = lane >> 4;
  const int lm   = lane & 15;
  const int qb   = blockIdx.x * 16;
  const int h    = blockIdx.y;
  const int b    = blockIdx.z;

  const float* Qp = Q  + ((size_t)b * Tq) * D + h * DH;
  const float* Kp = Kt + ((size_t)b * Tk) * D + h * DH;
  const float* Vp = Vt + ((size_t)b * Tk) * D + h * DH;
  float*       Op = O  + ((size_t)b * Tq) * D + h * DH;

  __shared__ _Float16 pls[16][34];

  // Resident Q fragments: two 16x32 chunks over dh, vectorized loads.
  const float* qrow = Qp + (size_t)(qb + lm) * D;
  v16h qa[2];
  qa[0] = load_row_frag(qrow, hlf);
  qa[1] = load_row_frag(qrow + 32, hlf);

  const v8f zero8 = {0.f, 0.f, 0.f, 0.f, 0.f, 0.f, 0.f, 0.f};
  v8f oacc[4] = {zero8, zero8, zero8, zero8};
  float mrow[8], lrow[8];
#pragma unroll
  for (int i = 0; i < 8; ++i) { mrow[i] = -1e30f; lrow[i] = 0.f; }

  const float scale = 0.125f;  // 1/sqrt(64)
  const int ktEnd = causal ? (qb + 16) : Tk;

  for (int kt = 0; kt < ktEnd; kt += 32) {
    // ---- scores: S[16q x 32k] as two 16x16 C-tiles ----
    const float* krow0 = Kp + (size_t)(kt + lm) * D;
    const float* krow1 = Kp + (size_t)(kt + 16 + lm) * D;
    v8f s0 = zero8, s1 = zero8;
#pragma unroll
    for (int c = 0; c < 2; ++c) {
      v16h kb0 = load_row_frag(krow0 + 32 * c, hlf);
      v16h kb1 = load_row_frag(krow1 + 32 * c, hlf);
      s0 = WMMA_F32_F16(qa[c], kb0, s0);
      s1 = WMMA_F32_F16(qa[c], kb1, s1);
    }

    // ---- online softmax (row m = i + 8*hlf, cols across 16 lanes) ----
    float p0[8], p1[8];
#pragma unroll
    for (int i = 0; i < 8; ++i) {
      float a0 = s0[i] * scale;
      float a1 = s1[i] * scale;
      if (causal) {
        int qidx = qb + i + 8 * hlf;
        a0 = (kt + lm > qidx)      ? -1e30f : a0;
        a1 = (kt + 16 + lm > qidx) ? -1e30f : a1;
      }
      float mcur = fmaxf(a0, a1);
#pragma unroll
      for (int x = 1; x < 16; x <<= 1) mcur = fmaxf(mcur, __shfl_xor(mcur, x, 32));
      float mnew = fmaxf(mrow[i], mcur);
      float corr = __expf(mrow[i] - mnew);
      float e0 = __expf(a0 - mnew);
      float e1 = __expf(a1 - mnew);
      float rs = e0 + e1;
#pragma unroll
      for (int x = 1; x < 16; x <<= 1) rs += __shfl_xor(rs, x, 32);
      lrow[i] = lrow[i] * corr + rs;
      mrow[i] = mnew;
      p0[i] = e0; p1[i] = e1;
#pragma unroll
      for (int nt = 0; nt < 4; ++nt) oacc[nt][i] *= corr;
    }

    // ---- re-layout P (C layout) -> A fragment via LDS ----
#pragma unroll
    for (int i = 0; i < 8; ++i) {
      pls[i + 8 * hlf][lm]      = (_Float16)p0[i];
      pls[i + 8 * hlf][16 + lm] = (_Float16)p1[i];
    }
    __syncthreads();
    v16h pa;
#pragma unroll
    for (int j = 0; j < 8; ++j) {
      int kk = frag_k(j, hlf);
      pa[2 * j]     = pls[lm][kk];
      pa[2 * j + 1] = pls[lm][kk + 1];
    }
    __syncthreads();

    // ---- O += P[16x32] @ V[32 x 64], 4 n-tiles of 16 ----
#pragma unroll
    for (int nt = 0; nt < 4; ++nt) {
      v16h vb;
#pragma unroll
      for (int j = 0; j < 8; ++j) {
        int kk = frag_k(j, hlf);
        size_t r0 = (size_t)(kt + kk) * D;
        vb[2 * j]     = (_Float16)Vp[r0 + nt * 16 + lm];
        vb[2 * j + 1] = (_Float16)Vp[r0 + D + nt * 16 + lm];
      }
      oacc[nt] = WMMA_F32_F16(pa, vb, oacc[nt]);
    }
  }

#pragma unroll
  for (int i = 0; i < 8; ++i) {
    float inv = 1.f / lrow[i];
    int m = qb + i + 8 * hlf;
#pragma unroll
    for (int nt = 0; nt < 4; ++nt)
      Op[(size_t)m * D + nt * 16 + lm] = oacc[nt][i] * inv;
  }
}

// ---------------------------------------------------------------------------
// y = LayerNorm(x + r) * gamma + beta   (d <= 1024; one block per row)
// ---------------------------------------------------------------------------
__global__ __launch_bounds__(256)
void ln_res_kernel(const float* __restrict__ x, const float* __restrict__ r,
                   const float* __restrict__ gamma, const float* __restrict__ beta,
                   float* __restrict__ y, int d) {
  __shared__ float row[1024];
  __shared__ float part[256];
  const int tid = threadIdx.x;
  const size_t base = (size_t)blockIdx.x * d;
  float s = 0.f;
  for (int j = tid; j < d; j += 256) {
    float v = x[base + j] + (r ? r[base + j] : 0.f);
    row[j] = v;
    s += v;
  }
  part[tid] = s; __syncthreads();
  for (int st = 128; st > 0; st >>= 1) { if (tid < st) part[tid] += part[tid + st]; __syncthreads(); }
  float mean = part[0] / d;
  __syncthreads();
  float vs = 0.f;
  for (int j = tid; j < d; j += 256) { float t = row[j] - mean; vs += t * t; }
  part[tid] = vs; __syncthreads();
  for (int st = 128; st > 0; st >>= 1) { if (tid < st) part[tid] += part[tid + st]; __syncthreads(); }
  float rstd = rsqrtf(part[0] / d + 1e-5f);
  for (int j = tid; j < d; j += 256)
    y[base + j] = (row[j] - mean) * rstd * gamma[j] + beta[j];
}

// ---------------------------------------------------------------------------
// Small helpers
// ---------------------------------------------------------------------------
__global__ void embed_kernel(const int* __restrict__ bytes, const float* __restrict__ emb,
                             float* __restrict__ out, int d) {
  int row = blockIdx.x;
  const float* e = emb + (size_t)bytes[row] * d;
  float* o = out + (size_t)row * d;
  for (int j = threadIdx.x; j < d; j += blockDim.x) o[j] = e[j];
}

__global__ void ngram_gather_kernel(const int* __restrict__ bytes, const float* __restrict__ tab,
                                    float* __restrict__ G, int n, int S, int d) {
  int row = blockIdx.x;             // b*S + t
  int b = row / S, t = row % S;
  int ws = t - n / 2;
  float* o = G + (size_t)row * d;
  if (ws < 0 || ws + n > S) {
    for (int j = threadIdx.x; j < d; j += blockDim.x) o[j] = 0.f;
    return;
  }
  int hsh = 0;
  for (int j = 0; j < n; ++j) hsh = (hsh * 256 + bytes[b * S + ws + j]) % 20000;
  const float* e = tab + (size_t)hsh * d;
  for (int j = threadIdx.x; j < d; j += blockDim.x) o[j] = e[j];
}

__global__ void qvec_kernel(const float* __restrict__ pq, const float* __restrict__ Wq,
                            const float* __restrict__ bq, float* __restrict__ qh, int d) {
  int j = blockIdx.x * blockDim.x + threadIdx.x;
  if (j >= d) return;
  float s = bq[j];
  for (int i = 0; i < d; ++i) s += pq[i] * Wq[(size_t)i * d + j];
  qh[j] = s;
}

// one thread per (b*P + p, head): pool PATCH byte positions with a fixed query
__global__ void patch_pool_kernel(const float* __restrict__ qh, const float* __restrict__ Kb,
                                  const float* __restrict__ Vb, float* __restrict__ pooled,
                                  int total, int D, int heads) {
  int idx = blockIdx.x * blockDim.x + threadIdx.x;
  if (idx >= total) return;
  const int PATCHn = 8;
  int hd = idx % heads;
  int bp = idx / heads;
  int dh = D / heads;
  const float* q = qh + hd * dh;
  size_t rowbase = (size_t)bp * PATCHn;   // S = P*PATCH so b*S + p*PATCH = bp*PATCH
  float s[8];
  float mx = -1e30f;
  for (int l = 0; l < PATCHn; ++l) {
    const float* kr = Kb + (rowbase + l) * D + hd * dh;
    float acc = 0.f;
    for (int t = 0; t < dh; ++t) acc += q[t] * kr[t];
    s[l] = acc * 0.125f;                  // 1/sqrt(64)
    mx = fmaxf(mx, s[l]);
  }
  float sum = 0.f;
  for (int l = 0; l < PATCHn; ++l) { s[l] = __expf(s[l] - mx); sum += s[l]; }
  float inv = 1.f / sum;
  float* o = pooled + (size_t)bp * D + hd * dh;
  for (int t = 0; t < dh; ++t) {
    float acc = 0.f;
    for (int l = 0; l < PATCHn; ++l) acc += s[l] * Vb[(rowbase + l) * D + hd * dh + t];
    o[t] = acc * inv;
  }
}

__global__ void add_pos_kernel(float* __restrict__ g, const float* __restrict__ pos,
                               int P, int d) {
  int row = blockIdx.x;
  int p = row % P;
  for (int j = threadIdx.x; j < d; j += blockDim.x)
    g[(size_t)row * d + j] += pos[(size_t)p * d + j];
}

__global__ void repeat_rows_kernel(const float* __restrict__ in, float* __restrict__ out,
                                   int rep, int d) {
  int row = blockIdx.x;
  const float* s = in + (size_t)(row / rep) * d;
  float* o = out + (size_t)row * d;
  for (int j = threadIdx.x; j < d; j += blockDim.x) o[j] = s[j];
}

// ---------------------------------------------------------------------------
// Host orchestration
// ---------------------------------------------------------------------------
extern "C" void kernel_launch(void* const* d_in, const int* in_sizes, int n_in,
                              void* d_out, int out_size, void* d_ws, size_t ws_size,
                              hipStream_t stream) {
  (void)in_sizes; (void)n_in; (void)out_size; (void)ws_size;
  const int B = 4, S = 1024, P = 128, DE = 768, DG = 1024, DD = 768;
  const int HE = 12, HG = 16, HD = 12, VOCAB = 256;

  const int* bytes = (const int*)d_in[0];
  auto F = [&](int i) -> const float* { return (const float*)d_in[i]; };

  // ---- param indices (JAX pytree sorted-key flatten; inputs 0,1 first) ----
  // 2:Wout 3:bout 4..11:byte_attn{Wk,Wo,Wq,Wv,bk,bo,bq,bv} 12:byte_emb
  // 13:byte_pos_emb 14:d_Win 15:d_bin 16..67:d_layers[2] 68..83:enc_layer
  // 84:g_Win 85:g_bin 86..137:g_layers[2] 138:g_pos 139..146:ngram 147..154:patch_attn
  // 155:patch_query
  // dec layer base L: cross=L+0..7, ffn=L+8..11{W1,W2,b1,b2}, ln1=L+12(b),13(g),
  //                   ln2=L+14,15, ln3=L+16,17, self=L+18..25
  const int WOUT = 2, BOUT = 3, BATTN = 4, BEMB = 12, DWIN = 14, DBIN = 15;
  const int DLAY0 = 16, DLAY1 = 42, ENC = 68, GWIN = 84, GBIN = 85;
  const int GLAY0 = 86, GLAY1 = 112, GPOS = 138, NGW = 139, NGB = 140, NTAB = 141;
  const int PATT = 147, PQ = 155;

  // ---- workspace layout (floats) ----
  float* ws = (float*)d_ws;
  size_t off = 0;
  auto alloc = [&](size_t n) { float* p = ws + off; off += n; return p; };
  float* X    = alloc((size_t)4096 * 768);   // byte-level activations
  float* Yb   = alloc((size_t)4096 * 768);   // sublayer output / residual
  float* Qb   = alloc((size_t)4096 * 768);
  float* Kb   = alloc((size_t)4096 * 768);
  float* Vb   = alloc((size_t)4096 * 768);
  float* Ob   = alloc((size_t)4096 * 768);
  float* Gbuf = alloc((size_t)4096 * 768);   // ngram gather / pooled / dec tmp
  float* Hf   = alloc((size_t)4096 * 3072);  // FFN hidden (also 512x4096)
  float* PR   = alloc((size_t)512 * 768);    // patch_repr
  float* Gx   = alloc((size_t)512 * 1024);   // global activations
  float* Gmem = alloc((size_t)512 * 1024);   // global memory (initial g)
  float* PRd  = alloc((size_t)512 * 768);    // decoder patch memory
  float* qh   = alloc(768);                  // projected learned query

  auto gemm = [&](const float* A, const float* W, const float* bias, float* C,
                  int M, int N, int K, int flags) {
    dim3 grid((unsigned)(N / 128), (unsigned)(M / 64));
    gemm_wmma_kernel<<<grid, 256, 0, stream>>>(A, W, bias, C, M, N, K, flags);
  };
  auto flash = [&](const float* Q, const float* K, const float* V, float* O,
                   int Tq, int Tk, int D, int causal) {
    dim3 grid((unsigned)(Tq / 16), (unsigned)(D / 64), (unsigned)B);
    flash_attn_kernel<<<grid, 32, 0, stream>>>(Q, K, V, O, Tq, Tk, D, causal);
  };
  auto ln = [&](const float* x, const float* r, int lnb, float* y, int rows, int d) {
    ln_res_kernel<<<rows, 256, 0, stream>>>(x, r, F(lnb + 1), F(lnb + 0), y, d);
  };
  // mha base mb: Wk=+0 Wo=+1 Wq=+2 Wv=+3 bk=+4 bo=+5 bq=+6 bv=+7
  auto mha = [&](const float* qin, const float* kvin, int mb, int Tq, int Tk,
                 int d, int causal, float* out) {
    int rq = B * Tq, rk = B * Tk;
    gemm(qin,  F(mb + 2), F(mb + 6), Qb, rq, d, d, 0);
    gemm(kvin, F(mb + 0), F(mb + 4), Kb, rk, d, d, 0);
    gemm(kvin, F(mb + 3), F(mb + 7), Vb, rk, d, d, 0);
    flash(Qb, Kb, Vb, Ob, Tq, Tk, d, causal);
    gemm(Ob, F(mb + 1), F(mb + 5), out, rq, d, d, 0);
  };
  auto declayer = [&](float* x, const float* mem, int L, int Tq, int Tk, int d,
                      int causal) {
    int rq = B * Tq;
    mha(x, x,   L + 18, Tq, Tq, d, causal, Yb);
    ln(x, Yb, L + 12, x, rq, d);
    mha(x, mem, L + 0,  Tq, Tk, d, 0, Yb);
    ln(x, Yb, L + 14, x, rq, d);
    gemm(x,  F(L + 8), F(L + 10), Hf, rq, 4 * d, d, 1);     // relu
    gemm(Hf, F(L + 9), F(L + 11), Yb, rq, d, 4 * d, 0);
    ln(x, Yb, L + 16, x, rq, d);
  };

  // ============ LocalEncoder ============
  embed_kernel<<<B * S, 256, 0, stream>>>(bytes, F(BEMB), X, DE);
  for (int ni = 0; ni < 6; ++ni) {
    int n = 3 + ni;
    ngram_gather_kernel<<<B * S, 256, 0, stream>>>(bytes, F(NTAB + ni), Gbuf, n, S, DE);
    gemm(Gbuf, F(NGW) + (size_t)ni * DE * DE, ni == 0 ? F(NGB) : nullptr, X,
         B * S, DE, DE, 2 /*accumulate*/);
  }
  // encoder layer: attn=ENC, ffn=ENC+8{W1,W2,b1,b2}, ln1=ENC+12(b),13(g), ln2=ENC+14,15
  mha(X, X, ENC, S, S, DE, 0, Yb);
  ln(X, Yb, ENC + 12, X, B * S, DE);
  gemm(X,  F(ENC + 8), F(ENC + 10), Hf, B * S, 4 * DE, DE, 1);
  gemm(Hf, F(ENC + 9), F(ENC + 11), Yb, B * S, DE, 4 * DE, 0);
  ln(X, Yb, ENC + 14, X, B * S, DE);

  // ============ patch pooling ============
  qvec_kernel<<<3, 256, 0, stream>>>(F(PQ), F(PATT + 2), F(PATT + 6), qh, DE);
  gemm(X, F(PATT + 0), F(PATT + 4), Kb, B * S, DE, DE, 0);
  gemm(X, F(PATT + 3), F(PATT + 7), Vb, B * S, DE, DE, 0);
  {
    int total = B * P * HE;
    patch_pool_kernel<<<(total + 255) / 256, 256, 0, stream>>>(qh, Kb, Vb, Gbuf,
                                                               total, DE, HE);
  }
  gemm(Gbuf, F(PATT + 1), F(PATT + 5), PR, B * P, DE, DE, 0);

  // ============ GlobalTransformer ============
  gemm(PR, F(GWIN), F(GBIN), Gx, B * P, DG, DE, 0);
  add_pos_kernel<<<B * P, 256, 0, stream>>>(Gx, F(GPOS), P, DG);
  hipMemcpyAsync(Gmem, Gx, (size_t)B * P * DG * sizeof(float),
                 hipMemcpyDeviceToDevice, stream);
  declayer(Gx, Gmem, GLAY0, P, P, DG, 1);
  declayer(Gx, Gmem, GLAY1, P, P, DG, 1);

  // ============ LocalDecoder ============
  gemm(Gx, F(DWIN), F(DBIN), PRd, B * P, DD, DG, 0);
  // collapsed byte cross-attn: dec = (pr @ Wv + bv) @ Wo + bo, repeated per byte
  gemm(PRd, F(BATTN + 3), F(BATTN + 7), Ob, B * P, DD, DD, 0);
  gemm(Ob,  F(BATTN + 1), F(BATTN + 5), Gbuf, B * P, DD, DD, 0);
  repeat_rows_kernel<<<B * S, 256, 0, stream>>>(Gbuf, X, 8, DD);
  declayer(X, PRd, DLAY0, S, P, DD, 0);
  declayer(X, PRd, DLAY1, S, P, DD, 0);

  // logits
  gemm(X, F(WOUT), F(BOUT), (float*)d_out, B * S, VOCAB, DD, 0);

  // heads sanity (unused at runtime, silences warnings)
  (void)HG; (void)HD;
}
